// RelationalUpdate_39290360824133
// MI455X (gfx1250) — compile-verified, matrix operand order checked by reading, not run on hardware
//
#include <hip/hip_runtime.h>
#include <hip/hip_bf16.h>

#define IN_F   256
#define OUT_F  256
#define NREL   8
#define M_TILE 64      // edges per block (4 sub-tiles of 16)
#define A_STRIDE 264   // bf16 elems per A-panel row: 256 + 8 pad -> 4-bank shift/row

typedef __attribute__((ext_vector_type(16))) __bf16 v16bf;
typedef __attribute__((ext_vector_type(8)))  __bf16 v8bf;
typedef __attribute__((ext_vector_type(8)))  float  v8f;
typedef __attribute__((ext_vector_type(4)))  float  f32x4;

// ---------------- bucketing pre-passes ----------------

__global__ void zero_meta_kernel(int* counts, int* cursors) {
    int t = threadIdx.x;
    if (t < NREL) { counts[t] = 0; cursors[t] = 0; }
}

__global__ void fill_perm_kernel(int* __restrict__ perm, int P) {
    int i = blockIdx.x * 256 + threadIdx.x;
    if (i < P) perm[i] = -1;
}

__global__ void hist_kernel(const int* __restrict__ etypes, int* __restrict__ counts, int E) {
    __shared__ int lc[NREL];
    int t = threadIdx.x;
    if (t < NREL) lc[t] = 0;
    __syncthreads();
    int e = blockIdx.x * 256 + t;
    if (e < E) atomicAdd(&lc[etypes[e]], 1);
    __syncthreads();
    if (t < NREL && lc[t]) atomicAdd(&counts[t], lc[t]);
}

// Exclusive scan of counts padded to multiples of M_TILE -> every 64-row tile
// of the permutation array holds edges of exactly one relation.
__global__ void scan_kernel(const int* __restrict__ counts, int* __restrict__ poffsets) {
    if (threadIdx.x == 0) {
        int s = 0;
        for (int r = 0; r < NREL; ++r) {
            poffsets[r] = s;
            s += (counts[r] + (M_TILE - 1)) & ~(M_TILE - 1);
        }
    }
}

__global__ void scatter_kernel(const int* __restrict__ etypes,
                               const int* __restrict__ poffsets,
                               int* __restrict__ cursors,
                               int* __restrict__ perm, int E) {
    __shared__ int lcount[NREL];
    __shared__ int lbase[NREL];
    int t = threadIdx.x;
    if (t < NREL) lcount[t] = 0;
    __syncthreads();
    int e = blockIdx.x * 256 + t;
    int ty = 0, lpos = 0;
    if (e < E) { ty = etypes[e]; lpos = atomicAdd(&lcount[ty], 1); }
    __syncthreads();
    if (t < NREL) lbase[t] = lcount[t] ? atomicAdd(&cursors[t], lcount[t]) : 0;
    __syncthreads();
    if (e < E) perm[poffsets[ty] + lbase[ty] + lpos] = e;
}

// Split f32 weights into hi/lo bf16, stored FRAGMENT-READY:
//   idx = (((r*256 + n)*8 + ks)*2 + hl)*16 + j
//   j<8 : K = ks*32 + hl*8 + j         (WMMA 16-bit B operand element order)
//   j>=8: K = ks*32 + 16 + hl*8 + (j-8)
// Each lane's 16-element fragment is one contiguous 32B load, no packing VALU.
__global__ void convert_kernels_kernel(const float* __restrict__ Ksrc,
                                       __bf16* __restrict__ khi,
                                       __bf16* __restrict__ klo) {
    int idx = blockIdx.x * 256 + threadIdx.x;   // over NREL*256*256
    int j   = idx & 15;
    int hl  = (idx >> 4) & 1;
    int ks  = (idx >> 5) & 7;
    int n   = (idx >> 8) & 255;
    int r   = idx >> 16;
    int K   = ks * 32 + hl * 8 + (j < 8 ? j : 8 + j);
    float x = Ksrc[(r << 16) + (K << 8) + n];
    __bf16 h = (__bf16)x;
    khi[idx] = h;
    klo[idx] = (__bf16)(x - (float)h);
}

// ---------------- main GEMM over permuted (bucketed, padded) edges ----------
// block = 256 threads (8 waves). M-tile = 64 edges (4 sub-tiles of 16), one
// relation per tile. Each wave owns 32 output columns (two 16-wide N tiles),
// B fragments are reused across the 4 M sub-tiles (4x less B cache traffic).
// K = 256 in steps of 32. C += Ahi*Bhi + Ahi*Blo + Alo*Bhi (~f32 accuracy).

__global__ void __launch_bounds__(256)
rgcn_gemm_kernel(const float* __restrict__ nodes,
                 const int* __restrict__ senders,
                 const int* __restrict__ etypes,
                 const int* __restrict__ perm,
                 const __bf16* __restrict__ khi,
                 const __bf16* __restrict__ klo,
                 float* __restrict__ out) {
    // A panels, fragment-ready: elem = m*A_STRIDE + ks*32 + hl*16 + j
    __shared__ __bf16 lds_ahi[M_TILE * A_STRIDE];
    __shared__ __bf16 lds_alo[M_TILE * A_STRIDE];
    __shared__ int lds_eid[M_TILE];
    __shared__ int lds_snd[M_TILE];
    __shared__ int lds_r;

    const int t = threadIdx.x;

    // ---- stage tile metadata ----
    if (t < M_TILE) {
        int e = perm[blockIdx.x * M_TILE + t];
        lds_eid[t] = e;
        lds_snd[t] = (e >= 0) ? senders[e] : 0;
        if (t == 0) lds_r = -1;
    }
    __syncthreads();
    if (t < M_TILE) {
        int e = lds_eid[t];
        if (e >= 0) lds_r = etypes[e];   // all valid rows share one relation
    }

    // ---- gather 64 node rows, split to bf16 hi/lo, write fragment-ready ----
    {
        int m = t >> 2;                              // 4 threads per row
        const float* row = nodes + (size_t)lds_snd[m] * IN_F;
        #pragma unroll
        for (int c = 0; c < 4; ++c) {
            int seg = (t & 3) * 4 + c;               // 16-float chunk index
            const f32x4* src = (const f32x4*)(row + seg * 16);
            f32x4 x0 = src[0], x1 = src[1], x2 = src[2], x3 = src[3];
            float x[16] = { x0.x, x0.y, x0.z, x0.w, x1.x, x1.y, x1.z, x1.w,
                            x2.x, x2.y, x2.z, x2.w, x3.x, x3.y, x3.z, x3.w };
            v8bf h0, l0, h1, l1;         // K=seg*16+i -> hl0 ; K=seg*16+8+i -> hl1
            #pragma unroll
            for (int i = 0; i < 8; ++i) {
                __bf16 a = (__bf16)x[i];
                h0[i] = a; l0[i] = (__bf16)(x[i] - (float)a);
                __bf16 b = (__bf16)x[i + 8];
                h1[i] = b; l1[i] = (__bf16)(x[i + 8] - (float)b);
            }
            int base  = m * A_STRIDE + (seg >> 1) * 32;   // ks = seg/2
            int jbase = (seg & 1) * 8;
            *(v8bf*)&lds_ahi[base + jbase]      = h0;     // hl=0 chunk
            *(v8bf*)&lds_ahi[base + 16 + jbase] = h1;     // hl=1 chunk
            *(v8bf*)&lds_alo[base + jbase]      = l0;
            *(v8bf*)&lds_alo[base + 16 + jbase] = l1;
        }
    }
    __syncthreads();

    const int r = lds_r;
    if (r < 0) return;                   // overshoot tile: all padding

    const int lane = t & 31;
    const int wv   = t >> 5;
    const int hl   = lane >> 4;          // K-half select
    const int mrow = lane & 15;          // M row (A) / N column (B, C)
    const int n0   = wv * 32 + mrow;
    const int n1   = n0 + 16;

    const __bf16* bh = khi + ((size_t)r << 16);
    const __bf16* bl = klo + ((size_t)r << 16);

    v8f acc[8] = {};                     // [mt*2 + ntile]

    #pragma unroll
    for (int ks = 0; ks < 8; ++ks) {
        // B fragments: loaded once per K-step, reused by 4 M sub-tiles
        const size_t f0 = ((((size_t)n0 * 8) + ks) * 2 + hl) * 16;
        const size_t f1 = ((((size_t)n1 * 8) + ks) * 2 + hl) * 16;
        v16bf bhi0 = *(const v16bf*)(bh + f0);
        v16bf blo0 = *(const v16bf*)(bl + f0);
        v16bf bhi1 = *(const v16bf*)(bh + f1);
        v16bf blo1 = *(const v16bf*)(bl + f1);

        #pragma unroll
        for (int mt = 0; mt < 4; ++mt) {
            const int aoff = (mt * 16 + mrow) * A_STRIDE + ks * 32 + hl * 16;
            v16bf ahi = *(const v16bf*)&lds_ahi[aoff];
            v16bf alo = *(const v16bf*)&lds_alo[aoff];

            acc[mt*2+0] = __builtin_amdgcn_wmma_f32_16x16x32_bf16(false, ahi, false, bhi0, (short)0, acc[mt*2+0], false, false);
            acc[mt*2+0] = __builtin_amdgcn_wmma_f32_16x16x32_bf16(false, ahi, false, blo0, (short)0, acc[mt*2+0], false, false);
            acc[mt*2+0] = __builtin_amdgcn_wmma_f32_16x16x32_bf16(false, alo, false, bhi0, (short)0, acc[mt*2+0], false, false);

            acc[mt*2+1] = __builtin_amdgcn_wmma_f32_16x16x32_bf16(false, ahi, false, bhi1, (short)0, acc[mt*2+1], false, false);
            acc[mt*2+1] = __builtin_amdgcn_wmma_f32_16x16x32_bf16(false, ahi, false, blo1, (short)0, acc[mt*2+1], false, false);
            acc[mt*2+1] = __builtin_amdgcn_wmma_f32_16x16x32_bf16(false, alo, false, bhi1, (short)0, acc[mt*2+1], false, false);
        }
    }

    // ---- scatter C rows to messages[edge]; streaming NT stores ----
    #pragma unroll
    for (int mt = 0; mt < 4; ++mt) {
        #pragma unroll
        for (int v = 0; v < 8; ++v) {
            int m = mt * 16 + hl * 8 + v;   // lanes 0-15: M=v ; lanes 16-31: M=8+v
            int e = lds_eid[m];
            if (e >= 0) {
                __builtin_nontemporal_store(acc[mt*2+0][v], out + (size_t)e * OUT_F + n0);
                __builtin_nontemporal_store(acc[mt*2+1][v], out + (size_t)e * OUT_F + n1);
            }
        }
    }
}

// ---------------- launcher ----------------

extern "C" void kernel_launch(void* const* d_in, const int* in_sizes, int n_in,
                              void* d_out, int out_size, void* d_ws, size_t ws_size,
                              hipStream_t stream) {
    const float* nodes   = (const float*)d_in[0];
    const int*   senders = (const int*)d_in[1];
    const int*   etypes  = (const int*)d_in[2];
    const float* kern    = (const float*)d_in[3];
    float*       out     = (float*)d_out;
    const int E = in_sizes[1];
    const int P = ((E + (M_TILE - 1)) & ~(M_TILE - 1)) + NREL * M_TILE;  // padded length

    char* ws = (char*)d_ws;
    int* counts   = (int*)(ws + 0);
    int* poffsets = (int*)(ws + 64);
    int* cursors  = (int*)(ws + 128);
    int* perm     = (int*)(ws + 256);
    size_t perm_bytes = ((size_t)P * 4 + 255) & ~(size_t)255;
    __bf16* khi = (__bf16*)(ws + 256 + perm_bytes);
    __bf16* klo = khi + (size_t)NREL * IN_F * OUT_F;

    const int eb = (E + 255) / 256;

    zero_meta_kernel<<<1, 32, 0, stream>>>(counts, cursors);
    fill_perm_kernel<<<(P + 255) / 256, 256, 0, stream>>>(perm, P);
    hist_kernel<<<eb, 256, 0, stream>>>(etypes, counts, E);
    scan_kernel<<<1, 32, 0, stream>>>(counts, poffsets);
    scatter_kernel<<<eb, 256, 0, stream>>>(etypes, poffsets, cursors, perm, E);
    convert_kernels_kernel<<<(NREL * IN_F * OUT_F) / 256, 256, 0, stream>>>(kern, khi, klo);

    const int gb = P / M_TILE;                        // tiles incl. worst-case padding
    rgcn_gemm_kernel<<<gb, 256, 0, stream>>>(nodes, senders, etypes, perm, khi, klo, out);
}
